// X_EfficientAdditiveAttnetion_26379689132775
// MI455X (gfx1250) — matrix operand order repaired and checked by reading, not compiled, not run on hardware
//
#include <hip/hip_runtime.h>

typedef __attribute__((ext_vector_type(16))) _Float16 v16h;
typedef __attribute__((ext_vector_type(8)))  _Float16 v8h;
typedef __attribute__((ext_vector_type(8)))  float    v8f;

#define NTOK 32768   /* 8*4096 tokens */
#define SCALE_G 0.0625f

__device__ __forceinline__ v8f wmma16(v16h a, v16h b, v8f c) {
  return __builtin_amdgcn_wmma_f32_16x16x32_f16(false, a, false, b, (short)0, c,
                                                false, false);
}

// ---------------- weight swizzle: fp32 [512,N] row-major -> fp16 WMMA-B layout
// B frag (16x16x32 f16): lane = n%16 + (k%32<16 ? 0:16), half h = k%16,
// blob(kt,nt) = 32 lanes * 16 halves contiguous.
__global__ void ea_swz(const float* __restrict__ W, _Float16* __restrict__ dst, int N) {
  int idx = blockIdx.x * 256 + threadIdx.x;
  if (idx >= 512 * N) return;
  int k = idx / N, n = idx % N;
  int kt = k >> 5, kk = k & 31, nt = n >> 4, nn = n & 15;
  int lane = nn + ((kk >= 16) ? 16 : 0);
  int hidx = ((kt * (N >> 4) + nt) * 32 + lane) * 16 + (kk & 15);
  dst[hidx] = (_Float16)W[idx];
}

// ---------------- PiF = p_i @ f_w  (4 x [512,256], fp32, tiny tiled GEMM)
__global__ void ea_pif(const float* __restrict__ p1, const float* __restrict__ p2,
                       const float* __restrict__ p3, const float* __restrict__ p4,
                       const float* __restrict__ fw, float* __restrict__ P) {
  __shared__ float As[16][16], Bs[16][17];
  int tx = threadIdx.x, ty = threadIdx.y;
  int o = blockIdx.x * 16 + tx, m = blockIdx.y * 16 + ty;
  const float* pw = (blockIdx.z == 0) ? p1 : (blockIdx.z == 1) ? p2
                  : (blockIdx.z == 2) ? p3 : p4;
  float acc = 0.f;
  for (int kk = 0; kk < 32; ++kk) {
    As[ty][tx] = pw[m * 512 + kk * 16 + tx];
    Bs[ty][tx] = fw[(kk * 16 + ty) * 256 + o];
    __syncthreads();
#pragma unroll
    for (int e = 0; e < 16; ++e) acc += As[ty][e] * Bs[e][tx];
    __syncthreads();
  }
  P[blockIdx.z * 131072 + m * 256 + o] = acc;
}

// ---------------- c = (b1+b2+b3+b4) @ f_w + f_b
__global__ void ea_cvec(const float* __restrict__ b1, const float* __restrict__ b2,
                        const float* __restrict__ b3, const float* __restrict__ b4,
                        const float* __restrict__ fw, const float* __restrict__ fb,
                        float* __restrict__ c) {
  int o = threadIdx.x;
  float acc = fb[o];
  for (int k = 0; k < 512; ++k)
    acc += (b1[k] + b2[k] + b3[k] + b4[k]) * fw[k * 256 + o];
  c[o] = acc;
}

__global__ void ea_zero(float* p, int n) {
  int i = blockIdx.x * 64 + threadIdx.x;
  if (i < n) p[i] = 0.f;
}

// ---------------- projection GEMM: Y = l2norm_row(X @ W + bias), fp16 out
// tile 64(M) x 512(N), 8 waves (2m x 4n), K=512 in 16 steps of 32.
__global__ __launch_bounds__(256) void ea_proj(const float* __restrict__ X,
                                               const _Float16* __restrict__ Wswz,
                                               const float* __restrict__ bias,
                                               _Float16* __restrict__ Y) {
  __shared__ _Float16 ldsA[2048];  // 4 m-tiles, A-fragment swizzled
  __shared__ float ldsSsq[64];
  const int t = threadIdx.x;
  const int row0 = blockIdx.x * 64;
  const int w = t >> 5, lane = t & 31;
  const int wm = w >> 2, wn = w & 3;
  if (t < 64) ldsSsq[t] = 0.f;

  v8f acc[2][8] = {};

  const int srow = t >> 2;
  const int sk0 = (t & 3) * 8;
  const int sidx = (srow >> 4) * 512 +
                   ((srow & 15) + ((sk0 >= 16) ? 16 : 0)) * 16 +
                   (sk0 & 15);
  const float* xp = X + (size_t)(row0 + srow) * 512 + sk0;

  for (int kt = 0; kt < 16; ++kt) {
    float4 x0 = *(const float4*)(xp + kt * 32);
    float4 x1 = *(const float4*)(xp + kt * 32 + 4);
    v8h hv;
    hv[0] = (_Float16)x0.x; hv[1] = (_Float16)x0.y;
    hv[2] = (_Float16)x0.z; hv[3] = (_Float16)x0.w;
    hv[4] = (_Float16)x1.x; hv[5] = (_Float16)x1.y;
    hv[6] = (_Float16)x1.z; hv[7] = (_Float16)x1.w;
    __syncthreads();
    *(v8h*)&ldsA[sidx] = hv;
    __syncthreads();
    v16h a0 = *(const v16h*)&ldsA[(wm * 2 + 0) * 512 + lane * 16];
    v16h a1 = *(const v16h*)&ldsA[(wm * 2 + 1) * 512 + lane * 16];
    const _Float16* bp = Wswz + ((size_t)(kt * 32 + wn * 8) * 32 + lane) * 16;
#pragma unroll
    for (int j = 0; j < 8; ++j) {
      v16h bf = *(const v16h*)(bp + j * 512);
      acc[0][j] = wmma16(a0, bf, acc[0][j]);
      acc[1][j] = wmma16(a1, bf, acc[1][j]);
    }
  }
  // epilogue: + bias, row L2 norm over all 512 cols, fp16 store
  float bv[8];
#pragma unroll
  for (int j = 0; j < 8; ++j) bv[j] = bias[wn * 128 + j * 16 + (lane & 15)];
  float part[16];
#pragma unroll
  for (int i = 0; i < 2; ++i)
#pragma unroll
    for (int r = 0; r < 8; ++r) {
      float s = 0.f;
#pragma unroll
      for (int j = 0; j < 8; ++j) {
        acc[i][j][r] += bv[j];
        s += acc[i][j][r] * acc[i][j][r];
      }
      part[i * 8 + r] = s;
    }
#pragma unroll
  for (int p = 0; p < 16; ++p) {
    part[p] += __shfl_xor(part[p], 1, 32);
    part[p] += __shfl_xor(part[p], 2, 32);
    part[p] += __shfl_xor(part[p], 4, 32);
    part[p] += __shfl_xor(part[p], 8, 32);
  }
  if ((lane & 15) == 0) {
    int mo = (lane >> 4) * 8;
#pragma unroll
    for (int i = 0; i < 2; ++i)
      for (int r = 0; r < 8; ++r)
        atomicAdd(&ldsSsq[(wm * 2 + i) * 16 + r + mo], part[i * 8 + r]);
  }
  __syncthreads();
  const int mo2 = (lane >= 16) ? 8 : 0;
#pragma unroll
  for (int i = 0; i < 2; ++i)
    for (int r = 0; r < 8; ++r) {
      int M = (wm * 2 + i) * 16 + r + mo2;
      float inv = 1.0f / fmaxf(sqrtf(ldsSsq[M]), 1e-12f);
      _Float16* yp = Y + (size_t)(row0 + M) * 512 + wn * 128 + (lane & 15);
#pragma unroll
      for (int j = 0; j < 8; ++j) yp[j * 16] = (_Float16)(acc[i][j][r] * inv);
    }
}

// ---------------- gating: r = (q . wg)*SCALE per token, ssq per batch
__global__ __launch_bounds__(256) void ea_gate(const _Float16* __restrict__ qI,
    const _Float16* __restrict__ qM, const float* __restrict__ wgI,
    const float* __restrict__ wgM, float* __restrict__ rI, float* __restrict__ rM,
    float* __restrict__ ssq) {
  __shared__ float wg0[512], wg1[512];
  __shared__ float bs[2];
  int t = threadIdx.x, blk = blockIdx.x;
  if (t < 2) bs[t] = 0.f;
  for (int i = t; i < 512; i += 256) { wg0[i] = wgI[i]; wg1[i] = wgM[i]; }
  __syncthreads();
  int token = blk * 32 + (t >> 3);
  int l8 = t & 7;
  size_t base = (size_t)token * 512 + l8 * 64;
  float aI = 0.f, aM = 0.f;
  for (int c = 0; c < 8; ++c) {
    v8h qa = *(const v8h*)(qI + base + c * 8);
    v8h qb = *(const v8h*)(qM + base + c * 8);
#pragma unroll
    for (int e = 0; e < 8; ++e) {
      aI += (float)qa[e] * wg0[l8 * 64 + c * 8 + e];
      aM += (float)qb[e] * wg1[l8 * 64 + c * 8 + e];
    }
  }
  aI += __shfl_xor(aI, 1, 32); aI += __shfl_xor(aI, 2, 32); aI += __shfl_xor(aI, 4, 32);
  aM += __shfl_xor(aM, 1, 32); aM += __shfl_xor(aM, 2, 32); aM += __shfl_xor(aM, 4, 32);
  if (l8 == 0) {
    float vI = aI * SCALE_G, vM = aM * SCALE_G;
    rI[token] = vI; rM[token] = vM;
    atomicAdd(&bs[0], vI * vI);
    atomicAdd(&bs[1], vM * vM);
  }
  __syncthreads();
  int b = blk >> 7;
  if (t == 0) atomicAdd(&ssq[b], bs[0]);
  if (t == 1) atomicAdd(&ssq[8 + b], bs[1]);
}

// ---------------- pooled queries: G[sel][b,d] = (sum_n r[n] q[n,d]) / max(||r||,eps)
__global__ __launch_bounds__(256) void ea_pool(const _Float16* __restrict__ qI,
    const _Float16* __restrict__ qM, const float* __restrict__ rI,
    const float* __restrict__ rM, const float* __restrict__ ssq,
    float* __restrict__ G) {
  __shared__ float red[256];
  int t = threadIdx.x, blk = blockIdx.x;
  int b = blk & 7, chunk = (blk >> 3) & 7, sel = blk >> 6;
  const _Float16* q = sel ? qM : qI;
  const float* r = sel ? rM : rI;
  int d = chunk * 64 + (t & 63);
  int ng = t >> 6;
  float acc = 0.f;
  for (int n = ng; n < 4096; n += 4)
    acc += r[b * 4096 + n] * (float)q[((size_t)b * 4096 + n) * 512 + d];
  red[t] = acc;
  __syncthreads();
  if (t < 64) {
    float tot = red[t] + red[t + 64] + red[t + 128] + red[t + 192];
    float inv = 1.0f / fmaxf(sqrtf(ssq[sel * 8 + b]), 1e-12f);
    G[sel * 4096 + b * 512 + d] = tot * inv;
  }
}

// ---------------- U_I/U_M = G-weighted combos of PiF, stored fp16 B-swizzled
__global__ __launch_bounds__(256) void ea_buildU(const float* __restrict__ G,
    const float* __restrict__ PF, _Float16* __restrict__ UI,
    _Float16* __restrict__ UM) {
  int idx = blockIdx.x * 256 + threadIdx.x;
  if (idx >= 8 * 512 * 256) return;
  int b = idx >> 17, k = (idx >> 8) & 511, o = idx & 255;
  float gI = G[b * 512 + k], gM = G[4096 + b * 512 + k];
  float p1 = PF[0 * 131072 + k * 256 + o];
  float p2 = PF[1 * 131072 + k * 256 + o];
  float p3 = PF[2 * 131072 + k * 256 + o];
  float p4 = PF[3 * 131072 + k * 256 + o];
  int kt = k >> 5, kk = k & 31, nt = o >> 4, nn = o & 15;
  int hidx = ((kt * 16 + nt) * 32 + (nn + ((kk >= 16) ? 16 : 0))) * 16 + (kk & 15);
  size_t boff = (size_t)b * 131072;
  UI[boff + hidx] = (_Float16)(gI * p1 + gM * p4);
  UM[boff + hidx] = (_Float16)(gI * p2 + gM * p3);
}

// ---------------- final: out = kI@U_I[b] + kM@U_M[b] + (qI+qM)@FW + c
__global__ __launch_bounds__(256) void ea_final(const _Float16* __restrict__ kIb,
    const _Float16* __restrict__ kMb, const _Float16* __restrict__ qIb,
    const _Float16* __restrict__ qMb, const _Float16* __restrict__ UI,
    const _Float16* __restrict__ UM, const _Float16* __restrict__ FWs,
    const float* __restrict__ cvec, float* __restrict__ out) {
  __shared__ _Float16 ldsA[2048];
  int t = threadIdx.x;
  int row0 = blockIdx.x * 64;
  int b = row0 >> 12;
  int w = t >> 5, lane = t & 31;
  int wm = w >> 2, wn = w & 3;
  v8f acc[2][4] = {};
  int srow = t >> 2, sk0 = (t & 3) * 8;
  int sidx = (srow >> 4) * 512 +
             ((srow & 15) + ((sk0 >= 16) ? 16 : 0)) * 16 +
             (sk0 & 15);
  size_t abase = (size_t)(row0 + srow) * 512 + sk0;
  const _Float16* Bsrc[3] = { UI + (size_t)b * 131072, UM + (size_t)b * 131072, FWs };
  for (int s = 0; s < 3; ++s) {
    for (int kt = 0; kt < 16; ++kt) {
      size_t go = abase + kt * 32;
      v8h av;
      if (s == 0)      av = *(const v8h*)(kIb + go);
      else if (s == 1) av = *(const v8h*)(kMb + go);
      else {
        v8h a1 = *(const v8h*)(qIb + go);
        v8h a2 = *(const v8h*)(qMb + go);
        av = a1 + a2;
      }
      __syncthreads();
      *(v8h*)&ldsA[sidx] = av;
      __syncthreads();
      v16h a0 = *(const v16h*)&ldsA[(wm * 2 + 0) * 512 + lane * 16];
      v16h a1f = *(const v16h*)&ldsA[(wm * 2 + 1) * 512 + lane * 16];
      const _Float16* bp = Bsrc[s] + ((size_t)(kt * 16 + wn * 4) * 32 + lane) * 16;
#pragma unroll
      for (int j = 0; j < 4; ++j) {
        v16h bf = *(const v16h*)(bp + j * 512);
        acc[0][j] = wmma16(a0, bf, acc[0][j]);
        acc[1][j] = wmma16(a1f, bf, acc[1][j]);
      }
    }
  }
  float cv[4];
#pragma unroll
  for (int j = 0; j < 4; ++j) cv[j] = cvec[wn * 64 + j * 16 + (lane & 15)];
  int mo = (lane >= 16) ? 8 : 0;
#pragma unroll
  for (int i = 0; i < 2; ++i)
    for (int r = 0; r < 8; ++r) {
      int M = (wm * 2 + i) * 16 + r + mo;
      float* op = out + (size_t)(row0 + M) * 256 + wn * 64 + (lane & 15);
#pragma unroll
      for (int j = 0; j < 4; ++j) op[j * 16] = acc[i][j][r] + cv[j];
    }
}

extern "C" void kernel_launch(void* const* d_in, const int* in_sizes, int n_in,
                              void* d_out, int out_size, void* d_ws, size_t ws_size,
                              hipStream_t stream) {
  (void)in_sizes; (void)n_in; (void)out_size; (void)ws_size;
  const float* x_I  = (const float*)d_in[0];
  const float* x_M  = (const float*)d_in[1];
  const float* wq_I = (const float*)d_in[2];
  const float* bq_I = (const float*)d_in[3];
  const float* wk_I = (const float*)d_in[4];
  const float* bk_I = (const float*)d_in[5];
  const float* wq_M = (const float*)d_in[6];
  const float* bq_M = (const float*)d_in[7];
  const float* wk_M = (const float*)d_in[8];
  const float* bk_M = (const float*)d_in[9];
  const float* wg_I = (const float*)d_in[10];
  const float* wg_M = (const float*)d_in[11];
  const float* p1_w = (const float*)d_in[12];
  const float* p1_b = (const float*)d_in[13];
  const float* p2_w = (const float*)d_in[14];
  const float* p2_b = (const float*)d_in[15];
  const float* p3_w = (const float*)d_in[16];
  const float* p3_b = (const float*)d_in[17];
  const float* p4_w = (const float*)d_in[18];
  const float* p4_b = (const float*)d_in[19];
  const float* f_w  = (const float*)d_in[20];
  const float* f_b  = (const float*)d_in[21];
  float* out = (float*)d_out;

  char* wsp = (char*)d_ws;
  auto alloc = [&](size_t bytes) -> void* {
    void* p = (void*)wsp;
    wsp += (bytes + 255) & ~(size_t)255;
    return p;
  };
  _Float16* qI  = (_Float16*)alloc((size_t)NTOK * 512 * 2);
  _Float16* kI  = (_Float16*)alloc((size_t)NTOK * 512 * 2);
  _Float16* qM  = (_Float16*)alloc((size_t)NTOK * 512 * 2);
  _Float16* kM  = (_Float16*)alloc((size_t)NTOK * 512 * 2);
  _Float16* WqIs = (_Float16*)alloc(512 * 512 * 2);
  _Float16* WkIs = (_Float16*)alloc(512 * 512 * 2);
  _Float16* WqMs = (_Float16*)alloc(512 * 512 * 2);
  _Float16* WkMs = (_Float16*)alloc(512 * 512 * 2);
  _Float16* FWs  = (_Float16*)alloc(512 * 256 * 2);
  float* PF   = (float*)alloc(4 * 512 * 256 * 4);
  _Float16* UI = (_Float16*)alloc(8 * 512 * 256 * 2);
  _Float16* UM = (_Float16*)alloc(8 * 512 * 256 * 2);
  float* rI   = (float*)alloc(NTOK * 4);
  float* rM   = (float*)alloc(NTOK * 4);
  float* ssq  = (float*)alloc(16 * 4);
  float* G    = (float*)alloc(2 * 8 * 512 * 4);
  float* cvec = (float*)alloc(256 * 4);

  ea_zero<<<1, 64, 0, stream>>>(ssq, 16);
  ea_swz<<<1024, 256, 0, stream>>>(wq_I, WqIs, 512);
  ea_swz<<<1024, 256, 0, stream>>>(wk_I, WkIs, 512);
  ea_swz<<<1024, 256, 0, stream>>>(wq_M, WqMs, 512);
  ea_swz<<<1024, 256, 0, stream>>>(wk_M, WkMs, 512);
  ea_swz<<<512, 256, 0, stream>>>(f_w, FWs, 256);
  ea_pif<<<dim3(16, 32, 4), dim3(16, 16), 0, stream>>>(p1_w, p2_w, p3_w, p4_w, f_w, PF);
  ea_cvec<<<1, 256, 0, stream>>>(p1_b, p2_b, p3_b, p4_b, f_w, f_b, cvec);

  ea_proj<<<512, 256, 0, stream>>>(x_I, WqIs, bq_I, qI);
  ea_proj<<<512, 256, 0, stream>>>(x_I, WkIs, bk_I, kI);
  ea_proj<<<512, 256, 0, stream>>>(x_M, WqMs, bq_M, qM);
  ea_proj<<<512, 256, 0, stream>>>(x_M, WkMs, bk_M, kM);

  ea_gate<<<1024, 256, 0, stream>>>(qI, qM, wg_I, wg_M, rI, rM, ssq);
  ea_pool<<<128, 256, 0, stream>>>(qI, qM, rI, rM, ssq, G);
  ea_buildU<<<4096, 256, 0, stream>>>(G, PF, UI, UM);

  ea_final<<<512, 256, 0, stream>>>(kI, kM, qI, qM, UI, UM, FWs, cvec, out);
}
